// PatchRWKV_56959856280334
// MI455X (gfx1250) — matrix operand order repaired and checked by reading, not compiled, not run on hardware
//
#include <hip/hip_runtime.h>

// ---------------------------------------------------------------------------
// RWKV-v4 forward for gfx1250 (MI455X)
//   - one-time weight transpose+bf16 convert (Wt[N][K])
//   - TDM (tensor_load_to_lds) double-buffered bf16 WMMA GEMMs
//   - per-channel WKV scan
// ---------------------------------------------------------------------------

typedef __attribute__((ext_vector_type(16))) __bf16        v16bf;
typedef __attribute__((ext_vector_type(8)))  float         v8f;
typedef __attribute__((ext_vector_type(4)))  unsigned int  u32x4;
typedef __attribute__((ext_vector_type(8)))  unsigned int  u32x8;

constexpr int    kB  = 8;
constexpr int    kT  = 512;
constexpr int    kH  = 768;
constexpr int    kL  = 12;
constexpr int    kFF = 3072;
constexpr int    kM  = kB * kT;                 // 4096 rows
constexpr size_t kNH = (size_t)kM * kH;         // 3,145,728
constexpr size_t kNF = (size_t)kM * kFF;        // 12,582,912

__device__ __forceinline__ unsigned short f2bf(float f) {
  unsigned int u = __float_as_uint(f);
  u += 0x7FFFu + ((u >> 16) & 1u);              // round-to-nearest-even
  return (unsigned short)(u >> 16);
}

union FragAB { v16bf v; u32x4 u[2]; };
static_assert(sizeof(v16bf) == 32, "bf16 fragment size");

// ---------------------------------------------------------------------------
// GEMM: C[M,N] = A_bf16[M,K] * Wt_bf16[N,K]^T  via V_WMMA_F32_16X16X32_BF16
// Block tile 128x64, 8 waves x (32x32, 2x2 WMMA accums), BK=32.
// A/B tiles staged into ping-pong LDS buffers by the Tensor Data Mover
// (tensor_load_to_lds, TENSORcnt), pipelined one tile ahead.
// EPI: 0 store f32 | 1 sigmoid f32 | 2 relu^2 -> bf16 | 4 res+aux*x | 5 res+x
// ---------------------------------------------------------------------------
template <int EPI>
__global__ __launch_bounds__(256) void rwkv_gemm_wmma(
    const unsigned short* __restrict__ A, const unsigned short* __restrict__ Wt,
    void* __restrict__ Cout, const float* __restrict__ res,
    const float* __restrict__ aux, int M, int N, int K) {
  __shared__ __align__(16) unsigned short sA[2][128 * 32];  // [buf][row][k]
  __shared__ __align__(16) unsigned short sB[2][64 * 32];   // [buf][n][k]

  const int tid  = threadIdx.x;
  const int lane = tid & 31;
  const int wid  = tid >> 5;
  const int wm   = wid & 3;        // wave row group (4)
  const int wn   = wid >> 2;       // wave col group (2)
  const int rowBase = blockIdx.y * 128;
  const int colBase = blockIdx.x * 64;

  const v8f vzero = {0.f, 0.f, 0.f, 0.f, 0.f, 0.f, 0.f, 0.f};
  v8f acc[2][2];
#pragma unroll
  for (int i = 0; i < 2; ++i)
#pragma unroll
    for (int j = 0; j < 2; ++j) acc[i][j] = vzero;

  // ---- TDM descriptors (D#): group1 is loop-invariant, group0 per issue ----
  const unsigned ldsA = (unsigned)(size_t)&sA[0][0];  // low 32b of flat = LDS byte addr
  const unsigned ldsB = (unsigned)(size_t)&sB[0][0];
  const unsigned uk = (unsigned)K;

  auto mk_g1 = [&](unsigned tileRows, unsigned tensorRows) -> u32x8 {
    u32x8 g;
    g[0] = 0x00010000u;                          // workgroup_mask=0, data_size=1 (2B)
    g[1] = (uk & 0xFFFFu) << 16;                 // tensor_dim0[15:0]
    g[2] = (uk >> 16) | ((tensorRows & 0xFFFFu) << 16);   // td0 hi | td1 lo
    g[3] = (tensorRows >> 16) | (32u << 16);     // td1 hi | tile_dim0=32
    g[4] = tileRows;                             // tile_dim1 | tile_dim2=0
    g[5] = uk;                                   // tensor_dim0_stride[31:0]
    g[6] = 0u;                                   // stride0 hi | stride1 lo
    g[7] = 0u;                                   // stride1 hi
    return g;
  };
  const u32x8 g1A = mk_g1(128u, (unsigned)M);
  const u32x8 g1B = mk_g1(64u, (unsigned)N);

  auto tdm_issue = [&](int kk, int buf) {
    unsigned long long gaA =
        (unsigned long long)(size_t)A + (((size_t)rowBase * K + kk) << 1);
    u32x4 g0a = {1u, ldsA + (unsigned)buf * (128u * 32u * 2u), (unsigned)gaA,
                 (unsigned)(gaA >> 32) | 0x80000000u};   // type=2 ("image")
    asm volatile("tensor_load_to_lds %0, %1" ::"s"(g0a), "s"(g1A) : "memory");
    unsigned long long gaB =
        (unsigned long long)(size_t)Wt + (((size_t)colBase * K + kk) << 1);
    u32x4 g0b = {1u, ldsB + (unsigned)buf * (64u * 32u * 2u), (unsigned)gaB,
                 (unsigned)(gaB >> 32) | 0x80000000u};
    asm volatile("tensor_load_to_lds %0, %1" ::"s"(g0b), "s"(g1B) : "memory");
  };

  const int nt = K >> 5;
  if (wid == 0) tdm_issue(0, 0);   // prologue: tile 0 in flight

  const int r  = lane & 15;
  const int hi = lane >> 4;

  for (int t = 0; t < nt; ++t) {
    if (wid == 0) {
      if (t + 1 < nt) {
        tdm_issue((t + 1) << 5, (t + 1) & 1);
        __builtin_amdgcn_s_wait_tensorcnt(2);    // tile t's 2 TDM ops retired
      } else {
        __builtin_amdgcn_s_wait_tensorcnt(0);
      }
    }
    __syncthreads();                             // TDM data visible to all waves

    const unsigned short* tA = &sA[t & 1][0];
    const unsigned short* tB = &sB[t & 1][0];

    FragAB af[2], bfr[2];
#pragma unroll
    for (int i = 0; i < 2; ++i) {
      int row = wm * 32 + i * 16 + r;   // A: lane r holds row, K chunks by hi
      af[i].u[0] = *reinterpret_cast<const u32x4*>(&tA[row * 32 + hi * 8]);
      af[i].u[1] = *reinterpret_cast<const u32x4*>(&tA[row * 32 + 16 + hi * 8]);
    }
#pragma unroll
    for (int j = 0; j < 2; ++j) {
      int n = wn * 32 + j * 16 + r;     // B: lane r holds column, K chunks by hi
      bfr[j].u[0] = *reinterpret_cast<const u32x4*>(&tB[n * 32 + hi * 8]);
      bfr[j].u[1] = *reinterpret_cast<const u32x4*>(&tB[n * 32 + 16 + hi * 8]);
    }
#pragma unroll
    for (int i = 0; i < 2; ++i)
#pragma unroll
      for (int j = 0; j < 2; ++j)
        acc[i][j] = __builtin_amdgcn_wmma_f32_16x16x32_bf16(
            false, af[i].v, false, bfr[j].v, (short)0, acc[i][j], false, false);

    __syncthreads();                    // all reads done before buffer reuse
  }

  // ---- epilogue: C/D layout is col = lane&15, row = (lane>>4)*8 + e ----
  const int col = lane & 15;
  const int rg  = (lane >> 4) * 8;
#pragma unroll
  for (int i = 0; i < 2; ++i) {
#pragma unroll
    for (int j = 0; j < 2; ++j) {
      int gr = rowBase + wm * 32 + i * 16 + rg;
      int gc = colBase + wn * 32 + j * 16 + col;
#pragma unroll
      for (int e = 0; e < 8; ++e) {
        size_t idx = (size_t)(gr + e) * N + gc;
        float x = acc[i][j][e];
        if constexpr (EPI == 0) {
          reinterpret_cast<float*>(Cout)[idx] = x;
        } else if constexpr (EPI == 1) {
          reinterpret_cast<float*>(Cout)[idx] = 1.f / (1.f + __expf(-x));
        } else if constexpr (EPI == 2) {
          float tt = fmaxf(x, 0.f);
          reinterpret_cast<unsigned short*>(Cout)[idx] = f2bf(tt * tt);
        } else if constexpr (EPI == 4) {
          reinterpret_cast<float*>(Cout)[idx] = res[idx] + aux[idx] * x;
        } else {  // EPI == 5
          reinterpret_cast<float*>(Cout)[idx] = res[idx] + x;
        }
      }
    }
  }
}

// ---------------------------------------------------------------------------
// One-time weight prep: W[K,N] f32 -> Wt[N,K] bf16 (tiled transpose via LDS)
// ---------------------------------------------------------------------------
__global__ __launch_bounds__(256) void rwkv_wprep(const float* __restrict__ W,
                                                  unsigned short* __restrict__ Wt,
                                                  int K, int N) {
  __shared__ unsigned short sT[64][65];   // +1 pad against bank conflicts
  const int k0 = blockIdx.y * 64;
  const int n0 = blockIdx.x * 64;
  const int tid = threadIdx.x;
  const int rr = tid >> 2;           // 0..63
  const int c0 = (tid & 3) << 4;     // 0,16,32,48
  const float* src = W + (size_t)(k0 + rr) * N + n0 + c0;
#pragma unroll
  for (int j = 0; j < 16; ++j) sT[c0 + j][rr] = f2bf(src[j]);
  __syncthreads();
  unsigned short* dst = Wt + (size_t)(n0 + rr) * K + k0 + c0;
#pragma unroll
  for (int j = 0; j < 16; ++j) dst[j] = sT[rr][c0 + j];
}

// ---------------------------------------------------------------------------
// LayerNorm: one block per row of length 768 (3 elems / thread, 8 waves)
// ---------------------------------------------------------------------------
__global__ __launch_bounds__(256) void rwkv_ln(const float* __restrict__ x,
                                               const float* __restrict__ w,
                                               const float* __restrict__ b,
                                               float* __restrict__ y) {
  const int row = blockIdx.x;
  const int tid = threadIdx.x;
  const float* xr = x + (size_t)row * kH;
  float v0 = xr[tid], v1 = xr[tid + 256], v2 = xr[tid + 512];
  float s = v0 + v1 + v2;
  float q = v0 * v0 + v1 * v1 + v2 * v2;
#pragma unroll
  for (int off = 16; off > 0; off >>= 1) {
    s += __shfl_down(s, off);
    q += __shfl_down(q, off);
  }
  __shared__ float ss[8], sq[8];
  __shared__ float s_mean, s_rstd;
  const int lane = tid & 31, widx = tid >> 5;
  if (lane == 0) { ss[widx] = s; sq[widx] = q; }
  __syncthreads();
  if (tid == 0) {
    float S = 0.f, Q = 0.f;
#pragma unroll
    for (int i = 0; i < 8; ++i) { S += ss[i]; Q += sq[i]; }
    float m = S * (1.f / kH);
    float var = Q * (1.f / kH) - m * m;
    s_mean = m;
    s_rstd = rsqrtf(var + 1e-5f);
  }
  __syncthreads();
  const float m = s_mean, rs = s_rstd;
  float* yr = y + (size_t)row * kH;
  yr[tid]       = (v0 - m) * rs * w[tid]       + b[tid];
  yr[tid + 256] = (v1 - m) * rs * w[tid + 256] + b[tid + 256];
  yr[tid + 512] = (v2 - m) * rs * w[tid + 512] + b[tid + 512];
}

// ---------------------------------------------------------------------------
// Token-shift + mix -> bf16 GEMM inputs (xv optional for FFN path)
// ---------------------------------------------------------------------------
__global__ __launch_bounds__(256) void rwkv_mix(
    const float* __restrict__ hn, const float* __restrict__ mk,
    const float* __restrict__ mv, const float* __restrict__ mr,
    unsigned short* __restrict__ xk, unsigned short* __restrict__ xv,
    unsigned short* __restrict__ xr) {
  size_t i = (size_t)blockIdx.x * blockDim.x + threadIdx.x;
  int h = (int)(i % kH);
  int t = (int)((i / kH) % kT);
  float cur  = hn[i];
  float prev = (t == 0) ? 0.f : hn[i - kH];
  float a = mk[h];
  xk[i] = f2bf(cur * a + prev * (1.f - a));
  if (xv != nullptr) {
    float c = mv[h];
    xv[i] = f2bf(cur * c + prev * (1.f - c));
  }
  float d = mr[h];
  xr[i] = f2bf(cur * d + prev * (1.f - d));
}

// ---------------------------------------------------------------------------
// WKV scan: one thread per (b,h) channel; sequential over T; coalesced in h
// ---------------------------------------------------------------------------
__global__ __launch_bounds__(256) void rwkv_wkv(
    const float* __restrict__ k, const float* __restrict__ v,
    const float* __restrict__ td, const float* __restrict__ tf,
    float* __restrict__ out) {
  int gid = blockIdx.x * blockDim.x + threadIdx.x;   // 0 .. B*H-1
  int b = gid / kH;
  int h = gid % kH;
  float u = tf[h];
  float w = -__expf(td[h]);
  float num = 0.f, den = 0.f, mx = -1e38f;
  const float* kp = k + (size_t)b * kT * kH + h;
  const float* vp = v + (size_t)b * kT * kH + h;
  float* op       = out + (size_t)b * kT * kH + h;
  for (int t = 0; t < kT; ++t) {
    float kt = kp[(size_t)t * kH];
    float vt = vp[(size_t)t * kH];
    float a  = kt + u;
    float mo = fmaxf(mx, a);
    float e1 = __expf(mx - mo);
    float e2 = __expf(a - mo);
    op[(size_t)t * kH] = (e1 * num + e2 * vt) / (e1 * den + e2);
    float ms = fmaxf(mx + w, kt);
    float f1 = __expf(mx + w - ms);
    float f2 = __expf(kt - ms);
    num = f1 * num + f2 * vt;
    den = f1 * den + f2;
    mx  = ms;
  }
}

// ra = bf16(rec * att)
__global__ __launch_bounds__(256) void rwkv_ra(const float* __restrict__ rec,
                                               const float* __restrict__ att,
                                               unsigned short* __restrict__ ra) {
  size_t i = (size_t)blockIdx.x * blockDim.x + threadIdx.x;
  ra[i] = f2bf(rec[i] * att[i]);
}

// bf16 convert
__global__ __launch_bounds__(256) void rwkv_cvt(const float* __restrict__ src,
                                                unsigned short* __restrict__ dst) {
  size_t i = (size_t)blockIdx.x * blockDim.x + threadIdx.x;
  dst[i] = f2bf(src[i]);
}

// ---------------------------------------------------------------------------
extern "C" void kernel_launch(void* const* d_in, const int* in_sizes, int n_in,
                              void* d_out, int out_size, void* d_ws, size_t ws_size,
                              hipStream_t stream) {
  const float* x          = (const float*)d_in[0];
  const float* pre_ln_w   = (const float*)d_in[1];
  const float* pre_ln_b   = (const float*)d_in[2];
  const float* ln1_w      = (const float*)d_in[3];
  const float* ln1_b      = (const float*)d_in[4];
  const float* ln2_w      = (const float*)d_in[5];
  const float* ln2_b      = (const float*)d_in[6];
  const float* time_decay = (const float*)d_in[7];
  const float* time_first = (const float*)d_in[8];
  const float* att_mk     = (const float*)d_in[9];
  const float* att_mv     = (const float*)d_in[10];
  const float* att_mr     = (const float*)d_in[11];
  const float* Wk         = (const float*)d_in[12];
  const float* Wv         = (const float*)d_in[13];
  const float* Wr         = (const float*)d_in[14];
  const float* Wo         = (const float*)d_in[15];
  const float* ffn_mk     = (const float*)d_in[16];
  const float* ffn_mr     = (const float*)d_in[17];
  const float* Fk         = (const float*)d_in[18];
  const float* Fv         = (const float*)d_in[19];
  const float* Fr         = (const float*)d_in[20];
  const float* ln_out_w   = (const float*)d_in[21];
  const float* ln_out_b   = (const float*)d_in[22];
  const float* head_w     = (const float*)d_in[23];

  char* ws = (char*)d_ws;
  size_t off = 0;
  auto alloc = [&](size_t bytes) {
    void* p = ws + off;
    off = (off + bytes + 255) & ~(size_t)255;
    return p;
  };
  float* h   = (float*)alloc(kNH * 4);
  float* hn  = (float*)alloc(kNH * 4);
  float* key = (float*)alloc(kNH * 4);
  float* val = (float*)alloc(kNH * 4);
  float* rec = (float*)alloc(kNH * 4);
  float* att = (float*)alloc(kNH * 4);
  unsigned short* xk = (unsigned short*)alloc(kNH * 2);
  unsigned short* xv = (unsigned short*)alloc(kNH * 2);
  unsigned short* xr = (unsigned short*)alloc(kNH * 2);
  unsigned short* ra = (unsigned short*)alloc(kNH * 2);
  unsigned short* kf = (unsigned short*)alloc(kNF * 2);

  const size_t hh = (size_t)kH * kH;
  const size_t hf = (size_t)kH * kFF;
  const size_t layerW = 5 * hh + 2 * hf;
  unsigned short* wt = (unsigned short*)alloc((kL * layerW + hh) * 2);

  const dim3 blk(256);
  const dim3 gLN(kM);
  const dim3 gElem((unsigned)(kNH / 256));
  const dim3 gG768(kH / 64, kM / 128);     // (12, 32)
  const dim3 gG3072(kFF / 64, kM / 128);   // (48, 32)
  const dim3 gWKV((kB * kH) / 256);        // 24
  const dim3 gT_hh(kH / 64, kH / 64);      // (12, 12)
  const dim3 gT_hf(kFF / 64, kH / 64);     // Fk: K=H, N=FF -> (48, 12)
  const dim3 gT_fh(kH / 64, kFF / 64);     // Fv: K=FF, N=H -> (12, 48)

  // ---- one-time weight prep: f32 [K,N] -> bf16 [N,K] ----
  for (int i = 0; i < kL; ++i) {
    unsigned short* wl = wt + (size_t)i * layerW;
    rwkv_wprep<<<gT_hh, blk, 0, stream>>>(Wk + i * hh, wl + 0 * hh, kH, kH);
    rwkv_wprep<<<gT_hh, blk, 0, stream>>>(Wv + i * hh, wl + 1 * hh, kH, kH);
    rwkv_wprep<<<gT_hh, blk, 0, stream>>>(Wr + i * hh, wl + 2 * hh, kH, kH);
    rwkv_wprep<<<gT_hh, blk, 0, stream>>>(Wo + i * hh, wl + 3 * hh, kH, kH);
    rwkv_wprep<<<gT_hh, blk, 0, stream>>>(Fr + i * hh, wl + 4 * hh, kH, kH);
    rwkv_wprep<<<gT_hf, blk, 0, stream>>>(Fk + i * hf, wl + 5 * hh, kH, kFF);
    rwkv_wprep<<<gT_fh, blk, 0, stream>>>(Fv + i * hf, wl + 5 * hh + hf, kFF, kH);
  }
  unsigned short* wtHead = wt + (size_t)kL * layerW;
  rwkv_wprep<<<gT_hh, blk, 0, stream>>>(head_w, wtHead, kH, kH);

  hipMemcpyAsync(h, x, kNH * 4, hipMemcpyDeviceToDevice, stream);
  // block-0 pre-LN (in-place safe: row reads precede row writes within block)
  rwkv_ln<<<gLN, blk, 0, stream>>>(h, pre_ln_w, pre_ln_b, h);

  for (int i = 0; i < kL; ++i) {
    unsigned short* wl = wt + (size_t)i * layerW;
    const unsigned short* tWk = wl + 0 * hh;
    const unsigned short* tWv = wl + 1 * hh;
    const unsigned short* tWr = wl + 2 * hh;
    const unsigned short* tWo = wl + 3 * hh;
    const unsigned short* tFr = wl + 4 * hh;
    const unsigned short* tFk = wl + 5 * hh;
    const unsigned short* tFv = wl + 5 * hh + hf;

    // ---- time mixing ----
    rwkv_ln<<<gLN, blk, 0, stream>>>(h, ln1_w + i * kH, ln1_b + i * kH, hn);
    rwkv_mix<<<gElem, blk, 0, stream>>>(hn, att_mk + i * kH, att_mv + i * kH,
                                        att_mr + i * kH, xk, xv, xr);
    rwkv_gemm_wmma<0><<<gG768, blk, 0, stream>>>(xk, tWk, key, nullptr, nullptr, kM, kH, kH);
    rwkv_gemm_wmma<0><<<gG768, blk, 0, stream>>>(xv, tWv, val, nullptr, nullptr, kM, kH, kH);
    rwkv_gemm_wmma<1><<<gG768, blk, 0, stream>>>(xr, tWr, rec, nullptr, nullptr, kM, kH, kH);
    rwkv_wkv<<<gWKV, blk, 0, stream>>>(key, val, time_decay + i * kH, time_first + i * kH, att);
    rwkv_ra<<<gElem, blk, 0, stream>>>(rec, att, ra);
    rwkv_gemm_wmma<5><<<gG768, blk, 0, stream>>>(ra, tWo, h, h, nullptr, kM, kH, kH);

    // ---- channel mixing ----
    rwkv_ln<<<gLN, blk, 0, stream>>>(h, ln2_w + i * kH, ln2_b + i * kH, hn);
    rwkv_mix<<<gElem, blk, 0, stream>>>(hn, ffn_mk + i * kH, nullptr,
                                        ffn_mr + i * kH, xk, nullptr, xr);
    rwkv_gemm_wmma<2><<<gG3072, blk, 0, stream>>>(xk, tFk, kf, nullptr, nullptr, kM, kFF, kH);
    rwkv_gemm_wmma<1><<<gG768, blk, 0, stream>>>(xr, tFr, rec, nullptr, nullptr, kM, kH, kH);
    rwkv_gemm_wmma<4><<<gG768, blk, 0, stream>>>(kf, tFv, h, h, rec, kM, kH, kFF);
  }

  // ---- head ----
  rwkv_ln<<<gLN, blk, 0, stream>>>(h, ln_out_w, ln_out_b, hn);
  rwkv_cvt<<<gElem, blk, 0, stream>>>(hn, xk);
  rwkv_gemm_wmma<0><<<gG768, blk, 0, stream>>>(xk, wtHead, d_out, nullptr, nullptr, kM, kH, kH);
}
